// Head_50818053047058
// MI455X (gfx1250) — matrix operand order repaired and checked by reading, not compiled, not run on hardware
//
#include <hip/hip_runtime.h>
#include <hip/hip_bf16.h>
#include <math.h>

// ---------------------------------------------------------------------------
// CDNA5 (gfx1250) fused attention: bf16 WMMA everywhere, fp32 accumulate.
// Round 3: both K and V key-blocks are DMA'd into LDS with
// GLOBAL_LOAD_ASYNC_TO_LDS_B128 (ASYNCcnt, double-buffered), and all WMMA
// fragments are preloaded into arrays before the WMMA bursts so the
// scheduler can clause the loads and stagger the waits.
// ---------------------------------------------------------------------------

typedef __bf16 bf16;
typedef __attribute__((ext_vector_type(16))) __bf16 v16bf;
typedef __attribute__((ext_vector_type(8)))  float  v8f;

#define NHEAD 2
#define EMB   256
#define BATCH 32
#define NVARS 2048
#define HD    128                 // head dim
#define M_TOK (BATCH * NVARS)     // 65536 tokens

union FragBF {
    v16bf v;
    uint4 u[2];
    bf16  h[16];
};

__device__ __forceinline__ v8f wmma_bf16(v16bf a, v16bf b, v8f c) {
    // D = A(16x32 bf16) * B(32x16 bf16) + C(16x16 f32)
    return __builtin_amdgcn_wmma_f32_16x16x32_bf16(
        /*neg_a=*/false, a, /*neg_b=*/false, b,
        /*c_mod=*/(short)0, c, /*reuse_a=*/false, /*reuse_b=*/false);
}

// Async global->LDS copy of 16 bytes per lane (VFLAT GLOBAL_LOAD_ASYNC_TO_LDS,
// GV mode, tracked with ASYNCcnt). lds_off is the wave-relative LDS byte
// offset (= low 32 bits of the flat shared-aperture address).
__device__ __forceinline__ void async_copy_b128(const void* gptr, unsigned lds_off) {
    unsigned long long ga = (unsigned long long)(uintptr_t)gptr;
    asm volatile("global_load_async_to_lds_b128 %0, %1, off"
                 :: "v"(lds_off), "v"(ga) : "memory");
}
__device__ __forceinline__ void wait_async0() {
    asm volatile("s_wait_asynccnt 0" ::: "memory");
}

// ---------------------------------------------------------------------------
// fp32 -> bf16 conversion (vectorized 4-wide)
// ---------------------------------------------------------------------------
__global__ void cvt_f32_to_bf16(const float* __restrict__ src,
                                bf16* __restrict__ dst, int n4) {
    int i = blockIdx.x * blockDim.x + threadIdx.x;
    int stride = gridDim.x * blockDim.x;
    for (; i < n4; i += stride) {
        float4 f = reinterpret_cast<const float4*>(src)[i];
        union { bf16 h[4]; unsigned long long ll; } pk;
        pk.h[0] = (bf16)f.x; pk.h[1] = (bf16)f.y;
        pk.h[2] = (bf16)f.z; pk.h[3] = (bf16)f.w;
        reinterpret_cast<unsigned long long*>(dst)[i] = pk.ll;
    }
}

// ---------------------------------------------------------------------------
// Generic bf16 GEMM: C[M, NDIM] = A[M, 256] * B[256, NDIM]
// Workgroup tile 128x128, K-step 32, 8 waves each owning a 64x32 subtile
// (4x2 grid of 16x16 WMMA accumulators).
// ---------------------------------------------------------------------------
template <int NDIM, bool OUT_BF16>
__global__ __launch_bounds__(256) void gemm_bf16_kernel(
        const bf16* __restrict__ A, const bf16* __restrict__ B,
        void* __restrict__ C) {
    __shared__ __align__(16) bf16 lds_a[128 * 32];   // [m][k], row-major
    __shared__ __align__(16) bf16 lds_bt[128 * 32];  // [n][k], B transposed

    const int tid  = threadIdx.x;
    const int lane = tid & 31;
    const int w    = tid >> 5;       // wave 0..7
    const int wm   = w >> 2;         // 0..1
    const int wn   = w & 3;          // 0..3
    const int half = lane >> 4;      // 0..1
    const int lc   = lane & 15;      // 0..15
    const int mBase = blockIdx.x * 128;
    const int nBase = blockIdx.y * 128;

    v8f acc[4][2];
    #pragma unroll
    for (int i = 0; i < 4; ++i)
        #pragma unroll
        for (int j = 0; j < 2; ++j)
            #pragma unroll
            for (int e = 0; e < 8; ++e) acc[i][j][e] = 0.f;

    for (int kt = 0; kt < 8; ++kt) {
        const int k0 = kt * 32;
        __syncthreads();
        {   // stage A tile (128x32): thread -> 16 contiguous bf16
            int row = tid >> 1, seg = tid & 1;
            const uint4* g = reinterpret_cast<const uint4*>(
                A + (size_t)(mBase + row) * 256 + k0 + seg * 16);
            uint4 d0 = g[0], d1 = g[1];
            uint4* s = reinterpret_cast<uint4*>(&lds_a[row * 32 + seg * 16]);
            s[0] = d0; s[1] = d1;
        }
        {   // stage B tile transposed (Bt[n][k])
            int kl = tid >> 3, seg = tid & 7;
            FragBF t;
            const uint4* g = reinterpret_cast<const uint4*>(
                B + (size_t)(k0 + kl) * NDIM + nBase + seg * 16);
            t.u[0] = g[0]; t.u[1] = g[1];
            #pragma unroll
            for (int e = 0; e < 16; ++e)
                lds_bt[(seg * 16 + e) * 32 + kl] = t.h[e];
        }
        __syncthreads();

        if (kt < 7) {   // hint next A tile toward the caches
            int row = tid >> 1, seg = tid & 1;
            __builtin_prefetch(A + (size_t)(mBase + row) * 256 + k0 + 32 + seg * 16, 0, 0);
        }

        // A fragments: lane lc owns row, halves split K chunks {c0, c0+16}
        FragBF af[4], bfr[2];
        #pragma unroll
        for (int i = 0; i < 4; ++i) {
            int r = wm * 64 + i * 16 + lc;
            af[i].u[0] = *reinterpret_cast<const uint4*>(&lds_a[r * 32 + half * 8]);
            af[i].u[1] = *reinterpret_cast<const uint4*>(&lds_a[r * 32 + half * 8 + 16]);
        }
        // B fragments: lane lc owns column n, halves split K chunks {c0, c0+16}
        #pragma unroll
        for (int j = 0; j < 2; ++j) {
            int n = wn * 32 + j * 16 + lc;
            const uint4* s = reinterpret_cast<const uint4*>(&lds_bt[n * 32 + half * 8]);
            bfr[j].u[0] = s[0]; bfr[j].u[1] = s[2];
        }
        #pragma unroll
        for (int i = 0; i < 4; ++i)
            #pragma unroll
            for (int j = 0; j < 2; ++j)
                acc[i][j] = wmma_bf16(af[i].v, bfr[j].v, acc[i][j]);
    }

    // store (C/D layout: VGPR v -> row v + 8*half, lane lc -> column)
    #pragma unroll
    for (int i = 0; i < 4; ++i)
        #pragma unroll
        for (int j = 0; j < 2; ++j)
            #pragma unroll
            for (int v = 0; v < 8; ++v) {
                int m = mBase + wm * 64 + i * 16 + half * 8 + v;
                int c = nBase + wn * 32 + j * 16 + lc;
                if constexpr (OUT_BF16)
                    ((bf16*)C)[(size_t)m * NDIM + c] = (bf16)acc[i][j][v];
                else
                    ((float*)C)[(size_t)m * NDIM + c] = acc[i][j][v];
            }
}

// ---------------------------------------------------------------------------
// Flash attention over one (batch, head): online softmax over 32-key blocks.
// Grid: (B*H, N/128). Block: 256 threads (8 waves, 16 query rows per wave).
// qkv layout: [b*N + n][768] bf16, with Q at col h*128, K at 256+h*128,
// V at 512+h*128.
// K and V blocks are DMA'd into LDS (ASYNCcnt, double buffered): block j+1
// streams in while block j is consumed. K rows are consumed in-place (the
// WMMA B-fragment wants contiguous head-dim chunks); V is transposed once
// per block into lds_vt for the P@V B-fragments.
// ---------------------------------------------------------------------------
__global__ __launch_bounds__(256) void flash_attn_kernel(
        const bf16* __restrict__ qkv, bf16* __restrict__ attn,
        const int* __restrict__ iter_flag) {
    __shared__ __align__(16) bf16 lds_kraw[2][32 * HD]; // raw K rows, 2 x 8KB
    __shared__ __align__(16) bf16 lds_vraw[2][32 * HD]; // raw V rows, 2 x 8KB
    __shared__ __align__(16) bf16 lds_vt[HD * 32];      // V transposed [d][j]
    __shared__ __align__(16) bf16 lds_p[8][16 * 32];    // per-wave P (16x32)

    const int tid  = threadIdx.x;
    const int lane = tid & 31;
    const int w    = tid >> 5;
    const int half = lane >> 4;
    const int lc   = lane & 15;
    const int bh   = blockIdx.x;       // 0..63
    const int b    = bh >> 1;
    const int h    = bh & 1;
    const int qBase = blockIdx.y * 128 + w * 16;   // this wave's 16 query rows
    const bool doMask = iter_flag[0] != 0;
    const float scale = 0.08838834764831845f;      // 1/sqrt(128)

    const size_t rs = 3 * EMB;  // 768 row stride in qkv
    const bf16* qb = qkv + (size_t)b * NVARS * rs + h * HD;
    const bf16* kb = qb + EMB;
    const bf16* vb = qb + 2 * EMB;

    // Q A-fragments (reused for every key block): 4 K-steps of 32 over hd=128
    FragBF qf[4];
    {
        const bf16* qp = qb + (size_t)(qBase + lc) * rs;
        #pragma unroll
        for (int ds = 0; ds < 4; ++ds) {
            int c0 = ds * 32 + half * 8;
            qf[ds].u[0] = *reinterpret_cast<const uint4*>(qp + c0);
            qf[ds].u[1] = *reinterpret_cast<const uint4*>(qp + c0 + 16);
        }
    }

    v8f acc_o[8];                    // 16 rows x 128 cols of O, f32
    #pragma unroll
    for (int t = 0; t < 8; ++t)
        #pragma unroll
        for (int e = 0; e < 8; ++e) acc_o[t][e] = 0.f;
    float m_i[8], l_i[8];
    #pragma unroll
    for (int v = 0; v < 8; ++v) { m_i[v] = -1e30f; l_i[v] = 0.f; }

    // issue async DMA of K and V key-block j0 into lds_{k,v}raw[buf]
    // (each thread copies 2 x 16B of K and 2 x 16B of V)
    auto issueKV = [&](int j0, int buf) {
        #pragma unroll
        for (int cc = 0; cc < 2; ++cc) {
            int c = tid + cc * 256;
            int jl = c >> 4, seg = c & 15;
            size_t rowoff = (size_t)(j0 + jl) * rs + seg * 8;
            async_copy_b128(kb + rowoff,
                (unsigned)(uintptr_t)&lds_kraw[buf][jl * HD + seg * 8]);
            async_copy_b128(vb + rowoff,
                (unsigned)(uintptr_t)&lds_vraw[buf][jl * HD + seg * 8]);
        }
    };

    issueKV(0, 0);

    for (int blk = 0; blk < NVARS / 32; ++blk) {
        const int j0 = blk * 32;
        const int buf = blk & 1;

        wait_async0();        // K,V(blk) landed in lds_*raw[buf]
        __syncthreads();      // ... and visible to all waves; lds_vt free

        if (blk + 1 < NVARS / 32) issueKV(j0 + 32, buf ^ 1); // overlap next DMA

        {   // transpose lds_vraw[buf] -> lds_vt[d][j]
            int jl = tid >> 3, seg = tid & 7;
            FragBF t;
            const uint4* s = reinterpret_cast<const uint4*>(
                &lds_vraw[buf][jl * HD + seg * 16]);
            t.u[0] = s[0]; t.u[1] = s[1];
            #pragma unroll
            for (int e = 0; e < 16; ++e)
                lds_vt[(seg * 16 + e) * 32 + jl] = t.h[e];
        }
        __syncthreads();

        // ---- S = Q K^T (16 x 32 per wave) ----
        // Preload all 8 K B-fragments (distinct regs -> clause + staggered waits)
        FragBF kf[8];
        #pragma unroll
        for (int ds = 0; ds < 4; ++ds) {
            int k0 = ds * 32 + half * 8;
            const uint4* p0 = reinterpret_cast<const uint4*>(
                &lds_kraw[buf][lc * HD + k0]);
            const uint4* p1 = reinterpret_cast<const uint4*>(
                &lds_kraw[buf][(16 + lc) * HD + k0]);
            kf[2 * ds].u[0]     = p0[0]; kf[2 * ds].u[1]     = p0[2];
            kf[2 * ds + 1].u[0] = p1[0]; kf[2 * ds + 1].u[1] = p1[2];
        }
        v8f s0, s1;
        #pragma unroll
        for (int e = 0; e < 8; ++e) { s0[e] = 0.f; s1[e] = 0.f; }
        #pragma unroll
        for (int ds = 0; ds < 4; ++ds) {
            s0 = wmma_bf16(qf[ds].v, kf[2 * ds].v, s0);
            s1 = wmma_bf16(qf[ds].v, kf[2 * ds + 1].v, s1);
        }

        // ---- scale + mask + online softmax (row = half*8 + v, 16 lanes) ----
        float alpha[8];
        #pragma unroll
        for (int v = 0; v < 8; ++v) {
            float e0 = s0[v] * scale, e1 = s1[v] * scale;
            if (doMask) {
                int qRow = qBase + half * 8 + v;
                if ((j0 + lc) == NVARS - 1 && qRow != NVARS - 1) e0 = -1e30f;
                if ((j0 + 16 + lc) == NVARS - 1 && qRow != NVARS - 1) e1 = -1e30f;
            }
            float mx = fmaxf(e0, e1);
            #pragma unroll
            for (int off = 1; off < 16; off <<= 1)
                mx = fmaxf(mx, __shfl_xor(mx, off, 32));
            float mn = fmaxf(m_i[v], mx);
            float p0 = __expf(e0 - mn);
            float p1 = __expf(e1 - mn);
            float rsum = p0 + p1;
            #pragma unroll
            for (int off = 1; off < 16; off <<= 1)
                rsum += __shfl_xor(rsum, off, 32);
            float al = __expf(m_i[v] - mn);
            l_i[v] = l_i[v] * al + rsum;
            m_i[v] = mn;
            alpha[v] = al;
            // spill P (C-layout) into per-wave LDS tile as bf16
            int rl = half * 8 + v;
            lds_p[w][rl * 32 + lc]      = (bf16)p0;
            lds_p[w][rl * 32 + 16 + lc] = (bf16)p1;
        }
        #pragma unroll
        for (int t = 0; t < 8; ++t)
            #pragma unroll
            for (int v = 0; v < 8; ++v) acc_o[t][v] *= alpha[v];

        // ---- O += P @ V_block : reload P in A-layout (in-wave DS ordering) ----
        FragBF pf;
        {
            int c0 = half * 8;
            pf.u[0] = *reinterpret_cast<const uint4*>(&lds_p[w][lc * 32 + c0]);
            pf.u[1] = *reinterpret_cast<const uint4*>(&lds_p[w][lc * 32 + c0 + 16]);
        }
        FragBF vf[8];
        #pragma unroll
        for (int t = 0; t < 8; ++t) {
            int d = t * 16 + lc, k0 = half * 8;
            const uint4* s = reinterpret_cast<const uint4*>(&lds_vt[d * 32 + k0]);
            vf[t].u[0] = s[0]; vf[t].u[1] = s[2];
        }
        #pragma unroll
        for (int t = 0; t < 8; ++t)
            acc_o[t] = wmma_bf16(pf.v, vf[t].v, acc_o[t]);
    }

    // ---- epilogue: O /= l, merge heads into [B*N, 256] bf16 ----
    #pragma unroll
    for (int t = 0; t < 8; ++t) {
        #pragma unroll
        for (int v = 0; v < 8; ++v) {
            int qRow = qBase + half * 8 + v;
            int d = t * 16 + lc;
            float o = acc_o[t][v] / l_i[v];
            attn[(size_t)(b * NVARS + qRow) * EMB + h * HD + d] = (bf16)o;
        }
    }
}

// ---------------------------------------------------------------------------
// Launch
// ---------------------------------------------------------------------------
extern "C" void kernel_launch(void* const* d_in, const int* in_sizes, int n_in,
                              void* d_out, int out_size, void* d_ws, size_t ws_size,
                              hipStream_t stream) {
    const float* x      = (const float*)d_in[0];
    const float* w_qkv  = (const float*)d_in[1];
    const float* w_proj = (const float*)d_in[2];
    const int*   iter   = (const int*)d_in[3];
    float* out = (float*)d_out;

    char* p = (char*)d_ws;
    bf16* xb   = (bf16*)p; p += (size_t)M_TOK * EMB * 2;        // 32 MB
    bf16* wqb  = (bf16*)p; p += (size_t)EMB * 3 * EMB * 2;      // 384 KB
    bf16* wpb  = (bf16*)p; p += (size_t)EMB * EMB * 2;          // 128 KB
    bf16* qkvb = (bf16*)p; p += (size_t)M_TOK * 3 * EMB * 2;    // 96 MB
    bf16* attn = (bf16*)p; p += (size_t)M_TOK * EMB * 2;        // 32 MB

    cvt_f32_to_bf16<<<dim3(4096), dim3(256), 0, stream>>>(x, xb, M_TOK * EMB / 4);
    cvt_f32_to_bf16<<<dim3(192),  dim3(256), 0, stream>>>(w_qkv, wqb, EMB * 3 * EMB / 4);
    cvt_f32_to_bf16<<<dim3(64),   dim3(256), 0, stream>>>(w_proj, wpb, EMB * EMB / 4);

    // qkv = x @ w_qkv  (65536 x 768)
    gemm_bf16_kernel<768, true><<<dim3(512, 6), dim3(256), 0, stream>>>(xb, wqb, (void*)qkvb);
    // fused attention -> [65536, 256] bf16
    flash_attn_kernel<<<dim3(BATCH * NHEAD, NVARS / 128), dim3(256), 0, stream>>>(qkvb, attn, iter);
    // out = attn @ w_proj (fp32 output)
    gemm_bf16_kernel<256, false><<<dim3(512, 2), dim3(256), 0, stream>>>(attn, wpb, (void*)out);
}
